// GraphConvOrganismClassifier_24756191494756
// MI455X (gfx1250) — matrix operand order repaired and checked by reading, not compiled, not run on hardware
//
#include <hip/hip_runtime.h>
#include <hip/hip_bf16.h>
#include <stdint.h>

// ---------------------------------------------------------------------------
// GraphConvOrganismClassifier for MI455X (gfx1250, wave32, WMMA)
// ---------------------------------------------------------------------------

typedef __attribute__((ext_vector_type(16))) __bf16 v16bf;
typedef __attribute__((ext_vector_type(8)))  __bf16 v8bf;
typedef __attribute__((ext_vector_type(8)))  float  v8f;

#define D_DIM 128
#define H_DIM 256
#define NCLS  6

// Set to 1 if the backend cannot select fptrunc f32->bf16 for gfx1250.
#define BF16_MANUAL_CVT 0

#if BF16_MANUAL_CVT
static __device__ __forceinline__ __bf16 f2bf(float f) {
    union { float f; uint32_t u; } in; in.f = f;
    uint32_t u = in.u;
    u += 0x7fffu + ((u >> 16) & 1u);
    union { uint16_t s; __bf16 b; } o; o.s = (uint16_t)(u >> 16);
    return o.b;
}
#else
static __device__ __forceinline__ __bf16 f2bf(float f) {
    return (__bf16)f;   // fptrunc, RNE — native bf16 convert on gfx1250
}
#endif

// ---------------------------------------------------------------------------
__global__ void k_zero(float* __restrict__ p, long long n) {
    long long i = (long long)blockIdx.x * blockDim.x + threadIdx.x;
    long long stride = (long long)gridDim.x * blockDim.x;
    for (; i < n; i += stride) p[i] = 0.0f;
}

__global__ void k_degree(const int* __restrict__ src, const int* __restrict__ dst,
                         float* __restrict__ dout, float* __restrict__ din, int e) {
    int i = blockIdx.x * blockDim.x + threadIdx.x;
    int stride = gridDim.x * blockDim.x;
    for (; i < e; i += stride) {
        atomicAdd(dout + src[i], 1.0f);
        atomicAdd(din + dst[i], 1.0f);
    }
}

__global__ void k_invsqrt(float* __restrict__ p, int n) {
    int i = blockIdx.x * blockDim.x + threadIdx.x;
    if (i < n) {
        float d = p[i];
        p[i] = (d > 0.0f) ? rsqrtf(d) : 1.0f;
    }
}

// One-shot: transpose + convert weight [R,C] f32 -> [C,R] bf16 so that the
// WMMA B-fragment (fixed N, contiguous K) is a single 32-byte vector load.
__global__ void k_cvt_transpose(const float* __restrict__ W, __bf16* __restrict__ WT,
                                int R, int C) {
    int i = blockIdx.x * blockDim.x + threadIdx.x;
    if (i < R * C) {
        int r = i / C, c = i % C;
        WT[(size_t)c * R + r] = f2bf(W[i]);
    }
}

// One wave per edge; lane handles 4 contiguous features. Scatter-add into agg,
// which at 134 MB is L2-resident (192 MB L2) -> atomics resolve in L2.
__global__ void k_scatter(const float* __restrict__ feat, const float* __restrict__ scale,
                          const int* __restrict__ src, const int* __restrict__ dst,
                          float* __restrict__ agg, int e) {
    int w = (int)(((unsigned)blockIdx.x * blockDim.x + threadIdx.x) >> 5);
    if (w >= e) return;
    int lane = threadIdx.x & 31;
    int s = src[w];
    int t = dst[w];
    float sc = scale ? scale[s] : 1.0f;
    const float4 v = *(const float4*)(feat + (size_t)s * D_DIM + lane * 4);
    float* o = agg + (size_t)t * D_DIM + lane * 4;
    atomicAdd(o + 0, v.x * sc);
    atomicAdd(o + 1, v.y * sc);
    atomicAdd(o + 2, v.z * sc);
    atomicAdd(o + 3, v.w * sc);
}

// ---------------------------------------------------------------------------
// GEMM1: h1 = relu((agg * c_in) @ W1 + b1), output bf16 [N, 256].
// One wave = 16 rows; K = 128 (4 WMMA steps of 32); sweeps 16 N-tiles.
// A-fragment k order used here: element e <-> K = kbA + e (kbA = 0 or 16),
// i.e. one contiguous 16-element run -> 4x float4 loads per K-step.
// B-fragment: element j <-> K = kbB + j at fixed N -> one v16bf load from W1T.
__global__ void k_gemm1(const float* __restrict__ agg, const float* __restrict__ cin,
                        const __bf16* __restrict__ W1T, const float* __restrict__ b1,
                        __bf16* __restrict__ h1) {
    int wave = blockIdx.x * (blockDim.x >> 5) + (threadIdx.x >> 5);
    int row0 = wave * 16;
    int lane = threadIdx.x & 31;
    int row  = row0 + (lane & 15);
    float ci = cin[row];
    int kbA  = (lane < 16) ? 0 : 16;
    int n16  = lane & 15;
    int kbB  = (lane < 16) ? 0 : 16;

    v16bf afr[4];
#pragma unroll
    for (int kk = 0; kk < 4; ++kk) {
        const float* ap = agg + (size_t)row * D_DIM + kk * 32;
        float4 c0 = *(const float4*)(ap + kbA);
        float4 c1 = *(const float4*)(ap + kbA + 4);
        float4 c2 = *(const float4*)(ap + kbA + 8);
        float4 c3 = *(const float4*)(ap + kbA + 12);
        afr[kk][0]  = f2bf(c0.x * ci); afr[kk][1]  = f2bf(c0.y * ci);
        afr[kk][2]  = f2bf(c0.z * ci); afr[kk][3]  = f2bf(c0.w * ci);
        afr[kk][4]  = f2bf(c1.x * ci); afr[kk][5]  = f2bf(c1.y * ci);
        afr[kk][6]  = f2bf(c1.z * ci); afr[kk][7]  = f2bf(c1.w * ci);
        afr[kk][8]  = f2bf(c2.x * ci); afr[kk][9]  = f2bf(c2.y * ci);
        afr[kk][10] = f2bf(c2.z * ci); afr[kk][11] = f2bf(c2.w * ci);
        afr[kk][12] = f2bf(c3.x * ci); afr[kk][13] = f2bf(c3.y * ci);
        afr[kk][14] = f2bf(c3.z * ci); afr[kk][15] = f2bf(c3.w * ci);
    }

    int mBase = row0 + ((lane < 16) ? 0 : 8);
    for (int nt = 0; nt < 16; ++nt) {
        int n = nt * 16 + n16;
        const __bf16* wcol = W1T + (size_t)n * D_DIM + kbB;  // W1T: [256,128] bf16
        v8f acc = {};
#pragma unroll
        for (int kk = 0; kk < 4; ++kk) {
            v16bf b = *(const v16bf*)(wcol + kk * 32);
            acc = __builtin_amdgcn_wmma_f32_16x16x32_bf16(
                false, afr[kk], false, b, (short)0, acc, false, false);
        }
        float bias = b1[n];
#pragma unroll
        for (int r = 0; r < 8; ++r) {
            float v = fmaxf(acc[r] + bias, 0.0f);  // bias + ReLU (v_max_num_f32)
            h1[(size_t)(mBase + r) * H_DIM + n] = f2bf(v);
        }
    }
}

// GEMM2: h2p = (h1 @ W2) * c_out, output f32 [N, 128].
// K = 256 (8 WMMA steps); sweeps 8 N-tiles. A is already bf16: two 16-byte
// v8bf loads merged with shufflevector; B is one v16bf load from W2T.
__global__ void k_gemm2(const __bf16* __restrict__ h1, const float* __restrict__ cout,
                        const __bf16* __restrict__ W2T, float* __restrict__ h2p) {
    int wave = blockIdx.x * (blockDim.x >> 5) + (threadIdx.x >> 5);
    int row0 = wave * 16;
    int lane = threadIdx.x & 31;
    int row  = row0 + (lane & 15);
    int kbA  = (lane < 16) ? 0 : 16;
    int n16  = lane & 15;
    int kbB  = (lane < 16) ? 0 : 16;

    v16bf afr[8];
#pragma unroll
    for (int kk = 0; kk < 8; ++kk) {
        const __bf16* ap = h1 + (size_t)row * H_DIM + kk * 32;
        v8bf lo = *(const v8bf*)(ap + kbA);
        v8bf hi = *(const v8bf*)(ap + kbA + 8);
        afr[kk] = __builtin_shufflevector(lo, hi,
                                          0, 1, 2, 3, 4, 5, 6, 7,
                                          8, 9, 10, 11, 12, 13, 14, 15);
    }

    int mBase = row0 + ((lane < 16) ? 0 : 8);
    float co[8];
#pragma unroll
    for (int r = 0; r < 8; ++r) co[r] = cout[mBase + r];

    for (int nt = 0; nt < 8; ++nt) {
        int n = nt * 16 + n16;
        const __bf16* wcol = W2T + (size_t)n * H_DIM + kbB;  // W2T: [128,256] bf16
        v8f acc = {};
#pragma unroll
        for (int kk = 0; kk < 8; ++kk) {
            v16bf b = *(const v16bf*)(wcol + kk * 32);
            acc = __builtin_amdgcn_wmma_f32_16x16x32_bf16(
                false, afr[kk], false, b, (short)0, acc, false, false);
        }
#pragma unroll
        for (int r = 0; r < 8; ++r)
            h2p[(size_t)(mBase + r) * D_DIM + n] = acc[r] * co[r];
    }
}

// ---------------------------------------------------------------------------
// Per-graph: mean over 128 nodes of (agg2 * c_in + b2), gather org embedding,
// then [x, o] @ Wc + bc -> 6 logits. One block (128 threads) per graph.
__global__ void k_final(const float* __restrict__ agg2, const float* __restrict__ cin,
                        const float* __restrict__ b2, const float* __restrict__ org_emb,
                        const int* __restrict__ org, const float* __restrict__ Wc,
                        const float* __restrict__ bc, float* __restrict__ out, int L) {
    __shared__ float xs[D_DIM];
    __shared__ float os[D_DIM];
    __shared__ float ci[D_DIM];
    int b = blockIdx.x;
    int d = threadIdx.x;

    ci[d] = cin[(size_t)b * L + d];   // L == 128 == blockDim.x
    __syncthreads();

    const float* base = agg2 + (size_t)b * L * D_DIM;
    float s = 0.0f;
    for (int l = 0; l < L; ++l)
        s += base[(size_t)l * D_DIM + d] * ci[l];
    xs[d] = s / (float)L + b2[d];
    os[d] = org_emb[(size_t)org[b] * D_DIM + d];
    __syncthreads();

    if (d < NCLS) {
        float acc = bc[d];
        for (int t = 0; t < D_DIM; ++t)
            acc += xs[t] * Wc[t * NCLS + d] + os[t] * Wc[(D_DIM + t) * NCLS + d];
        out[b * NCLS + d] = acc;
    }
}

// ---------------------------------------------------------------------------
extern "C" void kernel_launch(void* const* d_in, const int* in_sizes, int n_in,
                              void* d_out, int out_size, void* d_ws, size_t ws_size,
                              hipStream_t stream) {
    const float* n_feat  = (const float*)d_in[0];
    const float* W1      = (const float*)d_in[1];
    const float* b1      = (const float*)d_in[2];
    const float* W2      = (const float*)d_in[3];
    const float* b2      = (const float*)d_in[4];
    const float* org_emb = (const float*)d_in[5];
    const float* Wc      = (const float*)d_in[6];
    const float* bc      = (const float*)d_in[7];
    const int* edge_src  = (const int*)d_in[8];
    const int* edge_dst  = (const int*)d_in[9];
    const int* org       = (const int*)d_in[10];

    const int N = in_sizes[0] / D_DIM;   // 262144
    const int E = in_sizes[8];           // 2097152
    const int B = in_sizes[10];          // 2048
    const int L = N / B;                 // 128

    // Workspace carve-out (256B aligned)
    char* ws = (char*)d_ws;
    size_t off = 0;
    auto carve = [&](size_t bytes) {
        char* p = ws + off;
        off = (off + bytes + 255) & ~(size_t)255;
        return p;
    };
    float*  c_out = (float*)carve((size_t)N * 4);
    float*  c_in  = (float*)carve((size_t)N * 4);
    __bf16* W1T   = (__bf16*)carve((size_t)D_DIM * H_DIM * 2);
    __bf16* W2T   = (__bf16*)carve((size_t)D_DIM * H_DIM * 2);
    float*  agg   = (float*)carve((size_t)N * D_DIM * 4);
    float*  h2p   = (float*)carve((size_t)N * D_DIM * 4);
    __bf16* h1    = (__bf16*)carve((size_t)N * H_DIM * 2);
    (void)ws_size; (void)n_in; (void)out_size;

    // 0) weights -> bf16, transposed (tiny, one shot)
    k_cvt_transpose<<<(D_DIM * H_DIM + 255) / 256, 256, 0, stream>>>(W1, W1T, D_DIM, H_DIM);
    k_cvt_transpose<<<(D_DIM * H_DIM + 255) / 256, 256, 0, stream>>>(W2, W2T, H_DIM, D_DIM);

    // 1) degrees
    k_zero<<<1024, 256, 0, stream>>>(c_out, N);
    k_zero<<<1024, 256, 0, stream>>>(c_in, N);
    k_degree<<<2048, 256, 0, stream>>>(edge_src, edge_dst, c_out, c_in, E);
    k_invsqrt<<<(N + 255) / 256, 256, 0, stream>>>(c_out, N);
    k_invsqrt<<<(N + 255) / 256, 256, 0, stream>>>(c_in, N);

    // 2) layer 1: aggregate (n_feat * c_out) over edges, then WMMA GEMM + ReLU
    k_zero<<<4096, 256, 0, stream>>>(agg, (long long)N * D_DIM);
    k_scatter<<<(E + 7) / 8, 256, 0, stream>>>(n_feat, c_out, edge_src, edge_dst, agg, E);
    k_gemm1<<<N / 16 / 8, 256, 0, stream>>>(agg, c_in, W1T, b1, h1);

    // 3) layer 2: project first (WMMA), scale by c_out, then aggregate
    k_gemm2<<<N / 16 / 8, 256, 0, stream>>>(h1, c_out, W2T, h2p);
    k_zero<<<4096, 256, 0, stream>>>(agg, (long long)N * D_DIM);
    k_scatter<<<(E + 7) / 8, 256, 0, stream>>>(h2p, (const float*)nullptr,
                                               edge_src, edge_dst, agg, E);

    // 4) pool + classifier
    k_final<<<B, 128, 0, stream>>>(agg, c_in, b2, org_emb, org, Wc, bc,
                                   (float*)d_out, L);
}
